// SyntaxAwareAttention_68856915690002
// MI455X (gfx1250) — matrix operand order repaired and checked by reading, not compiled, not run on hardware
//
#include <hip/hip_runtime.h>
#include <hip/hip_bf16.h>
#include <stdint.h>

// Problem constants (fixed by the reference).
#define BB 4
#define SS 2048
#define HH 2048

typedef __attribute__((ext_vector_type(16))) __bf16 v16bf;
typedef __attribute__((ext_vector_type(8)))  __bf16 v8bf;
typedef __attribute__((ext_vector_type(4)))  __bf16 v4bf;
typedef __attribute__((ext_vector_type(8)))  float  v8f;
typedef __attribute__((ext_vector_type(4)))  unsigned int u32x4;
typedef __attribute__((ext_vector_type(8)))  int i32x8;
typedef __attribute__((ext_vector_type(4)))  int i32x4;

union Frag16 { v16bf v; v8bf h[2]; };

__device__ inline v8f wmma_bf16(v16bf a, v16bf b, v8f c) {
  // (neg_a, A, neg_b, B, c_mod, C, reuse_a, reuse_b)
  return __builtin_amdgcn_wmma_f32_16x16x32_bf16(false, a, false, b, (short)0, c,
                                                 false, false);
}

// ---------------------------------------------------------------------------
// Tensor Data Mover: 2D tile (bf16 elements) global -> LDS.
// Descriptor layout per CDNA5 ISA.
// ---------------------------------------------------------------------------
__device__ inline void tdm_load_2d_bf16(unsigned lds_byte, unsigned long long gaddr,
                                        unsigned tdim0, unsigned long long tdim1,
                                        unsigned tile0, unsigned tile1,
                                        unsigned long long stride0) {
  u32x4 g0;
  g0[0] = 1u;                                   // count=1, is_restore=0, no gather
  g0[1] = lds_byte;                             // lds_addr (bytes)
  g0[2] = (unsigned)(gaddr & 0xFFFFFFFFull);    // global_addr[31:0]
  g0[3] = (unsigned)((gaddr >> 32) & 0x01FFFFFFull) | 0x80000000u; // type=2
  i32x8 g1;
  g1[0] = (int)(1u << 16);                      // data_size=1 (2 bytes), mask=0
  g1[1] = (int)((tdim0 & 0xFFFFu) << 16);       // tensor_dim0[15:0] @ bit48
  g1[2] = (int)((tdim0 >> 16) | ((unsigned)(tdim1 & 0xFFFFull) << 16));
  g1[3] = (int)(((unsigned)(tdim1 >> 16) & 0xFFFFu) | (tile0 << 16));
  g1[4] = (int)(tile1 & 0xFFFFu);               // tile_dim1, tile_dim2=0
  g1[5] = (int)(stride0 & 0xFFFFFFFFull);       // tensor_dim0_stride[31:0]
  g1[6] = (int)((stride0 >> 32) & 0xFFFFull);   // stride hi, dim1_stride=0 (2D)
  g1[7] = 0;
  i32x4 z4 = {0, 0, 0, 0};
#if __clang_major__ >= 23
  i32x8 z8 = {0, 0, 0, 0, 0, 0, 0, 0};
  __builtin_amdgcn_tensor_load_to_lds(g0, g1, z4, z4, z8, 0);
#else
  __builtin_amdgcn_tensor_load_to_lds(g0, g1, z4, z4, 0);
#endif
}

// ---------------------------------------------------------------------------
// Kernel 1a: f32 -> bf16 elementwise convert (X).
// ---------------------------------------------------------------------------
__global__ __launch_bounds__(256) void cvt_f32_bf16(const float* __restrict__ in,
                                                    __bf16* __restrict__ out, int n) {
  int i = (blockIdx.x * 256 + threadIdx.x) * 4;
  if (i + 3 < n) {
    float4 f = *(const float4*)(in + i);
    v4bf o;
    o[0] = (__bf16)f.x; o[1] = (__bf16)f.y; o[2] = (__bf16)f.z; o[3] = (__bf16)f.w;
    *(v4bf*)(out + i) = o;
  }
}

// ---------------------------------------------------------------------------
// Kernel 1b: W[H][H] f32 -> WT[H][H] bf16 transposed (WT[n][k] = W[k][n]).
// 32x32 LDS tile so both global streams are coalesced.
// ---------------------------------------------------------------------------
__global__ __launch_bounds__(256) void transpose_bf16(const float* __restrict__ W,
                                                      __bf16* __restrict__ WT) {
  __shared__ float tile[32][33];
  int bx = blockIdx.x * 32, by = blockIdx.y * 32;
  int tx = threadIdx.x & 31, ty = threadIdx.x >> 5;
#pragma unroll
  for (int r = ty; r < 32; r += 8)
    tile[r][tx] = W[(size_t)(by + r) * HH + bx + tx];
  __syncthreads();
#pragma unroll
  for (int r = ty; r < 32; r += 8)
    WT[(size_t)(bx + r) * HH + by + tx] = (__bf16)tile[tx][r];
}

// ---------------------------------------------------------------------------
// Kernel 2: QKV projection GEMM.  grid = (HH/64, B*S/64, 3), block = 256.
// C[m][n] = sum_k X[m][k] * W[k][n] (+ bias, + syntax for K).
// A tile (64x32 bf16) staged via a DOUBLE-BUFFERED TDM pipeline: wave0 issues
// the DMA for tile i+1, then waits tensorcnt<=1 (TDM completes in order per
// wave) so tile i is guaranteed resident while tile i+1 overlaps the WMMAs.
// B fragments read contiguously from transposed bf16 weights.
// V is written transposed: Vt[b][h][s].
// ---------------------------------------------------------------------------
__global__ __launch_bounds__(256) void qkv_gemm(
    const __bf16* __restrict__ Xb,
    const __bf16* __restrict__ WqT, const __bf16* __restrict__ WkT,
    const __bf16* __restrict__ WvT,
    const float* __restrict__ bq, const float* __restrict__ bk,
    const float* __restrict__ bv,
    const int* __restrict__ styp, const float* __restrict__ semb,
    __bf16* __restrict__ Qb, __bf16* __restrict__ Kb, __bf16* __restrict__ Vt) {
  int z = blockIdx.z;
  const __bf16* WT = (z == 0) ? WqT : (z == 1) ? WkT : WvT;
  const float* bias = (z == 0) ? bq : (z == 1) ? bk : bv;

  __shared__ __bf16 As[2][64 * 32];  // 2 x 4 KB ping-pong A tiles
  int lane = threadIdx.x & 31, wave = threadIdx.x >> 5;
  int wm = wave & 3, wn = wave >> 2;
  int rowBase = blockIdx.y * 64;
  int colBase = blockIdx.x * 64 + wn * 32;

  v8f c0 = {0, 0, 0, 0, 0, 0, 0, 0};
  v8f c1 = {0, 0, 0, 0, 0, 0, 0, 0};
  unsigned ldsOff0 = (unsigned)(uintptr_t)(&As[0][0]);  // generic->LDS low 32b
  unsigned ldsOff1 = (unsigned)(uintptr_t)(&As[1][0]);
  unsigned long long gbase =
      (unsigned long long)(uintptr_t)Xb + (unsigned long long)rowBase * HH * 2ull;

  // Prologue: DMA tile 0 into buffer 0.
  if (threadIdx.x < 32) {
    tdm_load_2d_bf16(ldsOff0, gbase, HH, (unsigned long long)(BB * SS),
                     32, 64, HH);
  }

  for (int k0 = 0; k0 < HH; k0 += 32) {
    int cur = (k0 >> 5) & 1;
    __syncthreads();  // everyone done reading buf[cur^1] from previous step
    if (threadIdx.x < 32) {
      if (k0 + 32 < HH) {  // prefetch tile i+1 into the other buffer
        tdm_load_2d_bf16(cur ? ldsOff0 : ldsOff1,
                         gbase + (unsigned long long)(k0 + 32) * 2ull,
                         HH, (unsigned long long)(BB * SS), 32, 64, HH);
        __builtin_amdgcn_s_wait_tensorcnt(1);  // tile i resident; i+1 in flight
      } else {
        __builtin_amdgcn_s_wait_tensorcnt(0);  // last tile resident
      }
    }
    __syncthreads();  // tile cur ready for all waves

    const __bf16* Ab = &As[cur][0];
    int mloc = wm * 16 + (lane & 15);
    int kh = (lane >> 4) * 16;  // A: lanes16-31 carry K 16..31
    Frag16 a;
    a.h[0] = *(const v8bf*)(Ab + mloc * 32 + kh);
    a.h[1] = *(const v8bf*)(Ab + mloc * 32 + 8 + kh);

    int kB = k0 + (lane >> 4) * 16;  // B: lanes16-31 carry K 16..31
    int n0 = colBase + (lane & 15);
    v16bf b0v = *(const v16bf*)(WT + (size_t)n0 * HH + kB);
    v16bf b1v = *(const v16bf*)(WT + (size_t)(n0 + 16) * HH + kB);
    c0 = wmma_bf16(a.v, b0v, c0);
    c1 = wmma_bf16(a.v, b1v, c1);
  }

  int mBase = rowBase + wm * 16 + 8 * (lane >> 4);  // C: m = r + 8*(lane>=16)
#pragma unroll
  for (int sub = 0; sub < 2; ++sub) {
    v8f c = sub ? c1 : c0;
    int n = colBase + sub * 16 + (lane & 15);
    float bs = bias[n];
    if (z == 2) {  // V: store transposed, 8 consecutive s per lane -> 16B store
      int bI = rowBase / SS;
      int sl = (rowBase % SS) + wm * 16 + 8 * (lane >> 4);
      v8bf pk;
#pragma unroll
      for (int r = 0; r < 8; ++r) pk[r] = (__bf16)(c[r] + bs);
      *(v8bf*)(Vt + ((size_t)bI * HH + n) * SS + sl) = pk;
    } else if (z == 1) {  // K: + bias + 0.1*syntax_emb[type[s]]
#pragma unroll
      for (int r = 0; r < 8; ++r) {
        int m = mBase + r;
        float v = c[r] + bs + 0.1f * semb[(size_t)styp[m % SS] * HH + n];
        Kb[(size_t)m * HH + n] = (__bf16)v;
      }
    } else {  // Q
#pragma unroll
      for (int r = 0; r < 8; ++r)
        Qb[(size_t)(mBase + r) * HH + n] = (__bf16)(c[r] + bs);
    }
  }
}

// ---------------------------------------------------------------------------
// Kernel 3: attention.  grid = (S/16, B), block = 256 (8 waves).
// Dynamic LDS (CDNA5 WGP: 320KB):
//   sc: scores 16x2048 f32  (128KB)
//   pb: probs  16x2048 bf16 ( 64KB)
//   Qs: Q tile 16x2048 bf16 ( 64KB)  <- staged once via TDM, shared by 8 waves
// Wave w owns 16 column tiles {w*16 .. w*16+15} in both GEMM phases.
// ---------------------------------------------------------------------------
__global__ __launch_bounds__(256) void attn_kernel(
    const __bf16* __restrict__ Qb, const __bf16* __restrict__ Kb,
    const __bf16* __restrict__ Vt, const unsigned char* __restrict__ mask,
    float* __restrict__ ctx, float* __restrict__ probs) {
  extern __shared__ char smem[];
  float* sc = (float*)smem;                                        // [16][SS]
  __bf16* pb = (__bf16*)(smem + (size_t)16 * SS * sizeof(float));  // [16][SS]
  __bf16* Qs = pb + (size_t)16 * SS;                               // [16][SS]

  int b = blockIdx.y;
  int q0 = blockIdx.x * 16;
  int lane = threadIdx.x & 31, wave = threadIdx.x >> 5;
  const float scale = rsqrtf((float)HH);

  // Stage the whole 16-row Q tile into LDS with one TDM transfer; all 8 waves
  // then read A fragments from LDS instead of refetching Q 8x from global.
  if (threadIdx.x < 32) {
    unsigned long long ga = (unsigned long long)(uintptr_t)Qb +
                            ((unsigned long long)b * SS + q0) * HH * 2ull;
    tdm_load_2d_bf16((unsigned)(uintptr_t)Qs, ga, HH,
                     (unsigned long long)(BB * SS), /*tile0=*/HH, /*tile1=*/16,
                     /*stride0=*/HH);
    __builtin_amdgcn_s_wait_tensorcnt(0);
  }
  __syncthreads();

  // ---- Phase 1: scores = Q * K^T * scale ----
  {
    v8f c[16];
#pragma unroll
    for (int t = 0; t < 16; ++t)
#pragma unroll
      for (int e = 0; e < 8; ++e) c[t][e] = 0.0f;

    const __bf16* qrow = Qs + (size_t)(lane & 15) * HH;
    for (int k0 = 0; k0 < HH; k0 += 32) {
      int kA = k0 + ((lane >> 4) ? 16 : 0);
      Frag16 a;
      a.h[0] = *(const v8bf*)(qrow + kA);
      a.h[1] = *(const v8bf*)(qrow + kA + 8);
      int kB = k0 + (lane >> 4) * 16;
#pragma unroll
      for (int t = 0; t < 16; ++t) {
        int key = (wave * 16 + t) * 16 + (lane & 15);
        v16bf bv = *(const v16bf*)(Kb + ((size_t)b * SS + key) * HH + kB);
        c[t] = wmma_bf16(a.v, bv, c[t]);
      }
    }
#pragma unroll
    for (int t = 0; t < 16; ++t) {
      int col = (wave * 16 + t) * 16 + (lane & 15);
#pragma unroll
      for (int r = 0; r < 8; ++r) {
        int mr = r + 8 * (lane >> 4);
        sc[mr * SS + col] = c[t][r] * scale;
      }
    }
  }
  __syncthreads();

  // ---- Phase 2: mask + softmax; emit probs (f32 -> d_out, bf16 -> LDS) ----
  {
    int row = threadIdx.x >> 4;  // 16 threads per row, same half-wave
    int j = threadIdx.x & 15;
    const unsigned char* mrow = mask + ((size_t)b * SS + q0 + row) * SS;
    float mx = -3.0e38f;
    for (int col = j; col < SS; col += 16) {
      float s = sc[row * SS + col];
      if (mrow[col]) s = -1.0e9f;
      sc[row * SS + col] = s;
      mx = fmaxf(mx, s);
    }
#pragma unroll
    for (int off = 8; off > 0; off >>= 1) mx = fmaxf(mx, __shfl_xor(mx, off, 16));
    float sum = 0.0f;
    for (int col = j; col < SS; col += 16) {
      float e = __expf(sc[row * SS + col] - mx);
      sc[row * SS + col] = e;
      sum += e;
    }
#pragma unroll
    for (int off = 8; off > 0; off >>= 1) sum += __shfl_xor(sum, off, 16);
    float inv = 1.0f / sum;
    float* porow = probs + ((size_t)b * SS + q0 + row) * SS;
    for (int col = j; col < SS; col += 16) {
      float p = sc[row * SS + col] * inv;
      porow[col] = p;
      pb[row * SS + col] = (__bf16)p;
    }
  }
  __syncthreads();

  // ---- Phase 3: context = P * V  (B frag from transposed V, contiguous) ----
  {
    v8f c[16];
#pragma unroll
    for (int t = 0; t < 16; ++t)
#pragma unroll
      for (int e = 0; e < 8; ++e) c[t][e] = 0.0f;

    const __bf16* parow = pb + (size_t)(lane & 15) * SS;
    for (int k0 = 0; k0 < SS; k0 += 32) {
      int kA = k0 + ((lane >> 4) ? 16 : 0);
      Frag16 a;
      a.h[0] = *(const v8bf*)(parow + kA);
      a.h[1] = *(const v8bf*)(parow + kA + 8);
      int kB = k0 + (lane >> 4) * 16;
#pragma unroll
      for (int t = 0; t < 16; ++t) {
        int h = (wave * 16 + t) * 16 + (lane & 15);
        v16bf bv = *(const v16bf*)(Vt + ((size_t)b * HH + h) * SS + kB);
        c[t] = wmma_bf16(a.v, bv, c[t]);
      }
    }
#pragma unroll
    for (int t = 0; t < 16; ++t) {
      int h = (wave * 16 + t) * 16 + (lane & 15);
#pragma unroll
      for (int r = 0; r < 8; ++r) {
        int mr = r + 8 * (lane >> 4);
        ctx[((size_t)b * SS + q0 + mr) * HH + h] = c[t][r];
      }
    }
  }
}

// ---------------------------------------------------------------------------
extern "C" void kernel_launch(void* const* d_in, const int* in_sizes, int n_in,
                              void* d_out, int out_size, void* d_ws, size_t ws_size,
                              hipStream_t stream) {
  const float* X = (const float*)d_in[0];
  const int* styp = (const int*)d_in[1];
  const unsigned char* mask = (const unsigned char*)d_in[2];  // bool
  const float* Wq = (const float*)d_in[3];
  const float* bq = (const float*)d_in[4];
  const float* Wk = (const float*)d_in[5];
  const float* bk = (const float*)d_in[6];
  const float* Wv = (const float*)d_in[7];
  const float* bv = (const float*)d_in[8];
  const float* semb = (const float*)d_in[9];

  char* ws = (char*)d_ws;
  size_t off = 0;
  __bf16* Xb = (__bf16*)(ws + off);  off += (size_t)BB * SS * HH * 2;
  __bf16* WqT = (__bf16*)(ws + off); off += (size_t)HH * HH * 2;
  __bf16* WkT = (__bf16*)(ws + off); off += (size_t)HH * HH * 2;
  __bf16* WvT = (__bf16*)(ws + off); off += (size_t)HH * HH * 2;
  __bf16* Qb = (__bf16*)(ws + off);  off += (size_t)BB * SS * HH * 2;
  __bf16* Kb = (__bf16*)(ws + off);  off += (size_t)BB * SS * HH * 2;
  __bf16* Vt = (__bf16*)(ws + off);  off += (size_t)BB * HH * SS * 2;

  float* ctx = (float*)d_out;
  float* probs = ctx + (size_t)BB * SS * HH;

  // 1) bf16 conversions
  int nX = BB * SS * HH;
  cvt_f32_bf16<<<nX / (256 * 4), 256, 0, stream>>>(X, Xb, nX);
  dim3 tg(HH / 32, HH / 32);
  transpose_bf16<<<tg, 256, 0, stream>>>(Wq, WqT);
  transpose_bf16<<<tg, 256, 0, stream>>>(Wk, WkT);
  transpose_bf16<<<tg, 256, 0, stream>>>(Wv, WvT);

  // 2) QKV projections (z: 0=Q, 1=K, 2=V)
  qkv_gemm<<<dim3(HH / 64, (BB * SS) / 64, 3), 256, 0, stream>>>(
      Xb, WqT, WkT, WvT, bq, bk, bv, styp, semb, Qb, Kb, Vt);

  // 3) fused scores/softmax/context; 256KB dynamic LDS per WG (<= 320KB WGP)
  size_t smem = (size_t)16 * SS * sizeof(float) + (size_t)16 * SS * 2 +
                (size_t)16 * SS * 2;
  attn_kernel<<<dim3(SS / 16, BB), 256, smem, stream>>>(Qb, Kb, Vt, mask, ctx,
                                                        probs);
}
